// SoftRasterizer_35278861369367
// MI455X (gfx1250) — matrix operand order repaired and checked by reading, not compiled, not run on hardware
//
#include <hip/hip_runtime.h>

// SoftRasterizer via separable-Gaussian GEMM on CDNA5 WMMA.
// out[g] (128x128) = A (128x512) * B (512x128), A/B are 1-D Gaussian weights.
// Split-f16 (hi+lo) trick gives ~f32 accuracy with f32 WMMA accumulation.

typedef __attribute__((ext_vector_type(16))) _Float16 v16h;
typedef __attribute__((ext_vector_type(8)))  float    v8f;

#define BK_   16   // B*K point sets
#define N_    512  // points per set
#define H_    128
#define W_    128

__global__ __launch_bounds__(256)
void softraster_wmma(const float* __restrict__ pts, float* __restrict__ out) {
    __shared__ float s_px[N_];
    __shared__ float s_py[N_];

    const int g    = blockIdx.x;        // point-set index (b*K + k)
    const int m0   = blockIdx.y * 16;   // output row band
    const int tid  = threadIdx.x;
    const int lane = tid & 31;
    const int wave = tid >> 5;
    const int n0   = wave * 16;         // output column tile (1 per wave)

    // Stage this set's point coords into LDS (512 x float2 = 4 KB).
    const float* p = pts + (size_t)g * N_ * 2;
    for (int n = tid; n < N_; n += 256) {
        s_px[n] = p[2 * n + 0];
        s_py[n] = p[2 * n + 1];
    }
    __syncthreads();

    const float c      = 1.0f / (2.0f * (1.0f + 1e-8f));  // SIGMA = 1.0
    const int   hi16   = lane >> 4;      // lane half (0/1)
    const int   l15    = lane & 15;
    const float ycoord = (float)(m0 + l15);   // A-fragment row (M = lane&15)
    const float xcoord = (float)(n0 + l15);   // B-fragment col (N = lane&15)

    v8f acc = {};

    #pragma unroll 1
    for (int ks = 0; ks < N_; ks += 32) {
        v16h ahi, alo, bhi, blo;
        #pragma unroll
        for (int e = 0; e < 16; ++e) {
            // ---- A (16x32 f16): lane&15 = M; K = 8*(lane>=16) + (e&7) + 16*(e>=8)
            const int ka = (hi16 << 3) + (e & 7) + ((e >> 3) << 4);
            const float dy = ycoord - s_py[ks + ka];
            const float wa = __expf(-dy * dy * c);
            const _Float16 ha = (_Float16)wa;
            ahi[e] = ha;
            alo[e] = (_Float16)(wa - (float)ha);

            // ---- B (32x16 f16): lane&15 = N; K = 8*(lane>=16) + (e&7) + 16*(e>=8)
            const int kb = (hi16 << 3) + (e & 7) + ((e >> 3) << 4);
            const float dx = xcoord - s_px[ks + kb];
            const float wb = __expf(-dx * dx * c);
            const _Float16 hb = (_Float16)wb;
            bhi[e] = hb;
            blo[e] = (_Float16)(wb - (float)hb);
        }
        // acc += Ahi*Bhi + Ahi*Blo + Alo*Bhi  (drop lo*lo; ~2^-22 residual)
        acc = __builtin_amdgcn_wmma_f32_16x16x32_f16(false, ahi, false, bhi,
                                                     (short)0, acc, false, false);
        acc = __builtin_amdgcn_wmma_f32_16x16x32_f16(false, ahi, false, blo,
                                                     (short)0, acc, false, false);
        acc = __builtin_amdgcn_wmma_f32_16x16x32_f16(false, alo, false, bhi,
                                                     (short)0, acc, false, false);
    }

    // C/D layout: VGPR r -> M = r + 8*(lane>=16); N = lane&15.
    float* o = out + (size_t)g * H_ * W_;
    #pragma unroll
    for (int r = 0; r < 8; ++r) {
        const int row = m0 + r + (hi16 << 3);
        const int col = n0 + l15;
        o[row * W_ + col] = acc[r];
    }
}

extern "C" void kernel_launch(void* const* d_in, const int* in_sizes, int n_in,
                              void* d_out, int out_size, void* d_ws, size_t ws_size,
                              hipStream_t stream) {
    (void)in_sizes; (void)n_in; (void)d_ws; (void)ws_size; (void)out_size;
    const float* pts = (const float*)d_in[0];   // (B,K,N,2) f32
    float*       out = (float*)d_out;           // (B,K,H,W) f32

    dim3 grid(BK_, H_ / 16);   // 16 point sets x 8 row bands = 128 workgroups
    softraster_wmma<<<grid, 256, 0, stream>>>(pts, out);
}
